// SelfOrganizingBrain_81724637708794
// MI455X (gfx1250) — matrix operand order repaired and checked by reading, not compiled, not run on hardware
//
#include <hip/hip_runtime.h>
#include <hip/hip_bf16.h>

// ---------------------------------------------------------------------------
// SelfOrganizingBrain on MI455X (gfx1250): grouped bf16 WMMA GEMMs.
// IN=784 (padded to 800), E=512, BSZ=8, A=2 (T=64 blocks), J=7, NC=128, B=512.
// ---------------------------------------------------------------------------

#define BATCH   512
#define IN_DIM  784
#define IN_PAD  800         // 784 rounded up to multiple of 32 (zero padded)
#define EDIM    512
#define NC      128
#define NBLK    64          // T = 8^2 blocks
#define ABSZ    16          // A*BSZ logits
#define JUMPS   7
#define MAXTILES 96         // sum ceil(n_g/16) <= 64 + 512/16

typedef __attribute__((ext_vector_type(16))) __bf16 v16bf;
typedef __attribute__((ext_vector_type(8)))  float  v8f;
typedef __attribute__((ext_vector_type(4)))  float  v4f;

__device__ __forceinline__ v8f wmma_bf16(v16bf a, v16bf b, v8f c) {
    return __builtin_amdgcn_wmma_f32_16x16x32_bf16(
        false, a, false, b, (short)0, c, false, false);
}

// Pack 8 consecutive f32 (two b128 loads) into fragment elements e0..e0+7.
__device__ __forceinline__ void pack8(v16bf& a, int e0, const float* __restrict__ p) {
    v4f lo = *(const v4f*)p;
    v4f hi = *(const v4f*)(p + 4);
#pragma unroll
    for (int j = 0; j < 4; ++j) a[e0 + j] = (__bf16)lo[j];
#pragma unroll
    for (int j = 0; j < 4; ++j) a[e0 + 4 + j] = (__bf16)hi[j];
}

// A-matrix 16x32 bf16 fragment: lane half selects 16 contiguous K (matches B).
__device__ __forceinline__ v16bf load_a_frag(const float* __restrict__ Arow,
                                             int k0, int lane) {
    int kh = (lane >> 4) * 16;
    v16bf a;
    pack8(a, 0, Arow + k0 + kh);        // K = k0+kh   .. +7
    pack8(a, 8, Arow + k0 + kh + 8);    // K = k0+kh+8 .. +15
    return a;
}

// B-matrix 32x16 bf16 fragment: lane n = n&15 fixed column, 16 sequential K.
// Strided per lane, but lanes 0-15 cover 16 consecutive columns -> coalesced.
__device__ __forceinline__ v16bf load_b_frag(const float* __restrict__ W, int ldw,
                                             int k0, int n, int lane) {
    const float* p = W + (size_t)(k0 + (lane >> 4) * 16) * ldw + n;
    v16bf b;
#pragma unroll
    for (int e = 0; e < 16; ++e) b[e] = (__bf16)p[(size_t)e * ldw];
    return b;
}

// ---- Dense GEMM: C[M,N] = act(A[M,K] @ W[K,N] + bias), K % 32 == 0 ---------
// One wave per 16 x (16*NT) tile. All B fragments of a K-step are built
// before any WMMA issues (gives scheduler co-exec work for WMMA WAR slots).
template<int NT>
__global__ __launch_bounds__(32)
void dense_gemm_kernel(const float* __restrict__ A, const float* __restrict__ W,
                       const float* __restrict__ bias, float* __restrict__ C,
                       int N, int K, int relu) {
    int lane = threadIdx.x;
    int m0 = blockIdx.x * 16;
    int n0 = blockIdx.y * 16 * NT;
    const float* Arow = A + (size_t)(m0 + (lane & 15)) * K;
    int nl = lane & 15;
    v8f acc[NT];
#pragma unroll
    for (int t = 0; t < NT; ++t) acc[t] = {};

    for (int k0 = 0; k0 < K; k0 += 32) {
        if (k0 + 32 < K)
            __builtin_prefetch(W + (size_t)(k0 + 32) * N + n0, 0, 1);
        v16bf a = load_a_frag(Arow, k0, lane);
        v16bf b[NT];
#pragma unroll
        for (int t = 0; t < NT; ++t)
            b[t] = load_b_frag(W, N, k0, n0 + t * 16 + nl, lane);
#pragma unroll
        for (int t = 0; t < NT; ++t)
            acc[t] = wmma_bf16(a, b[t], acc[t]);
    }
#pragma unroll
    for (int t = 0; t < NT; ++t) {
        int n = n0 + t * 16 + nl;
        float bv = bias[n];
#pragma unroll
        for (int i = 0; i < 8; ++i) {
            int m = m0 + (lane >> 4) * 8 + i;
            float v = acc[t][i] + bv;
            if (relu) v = fmaxf(v, 0.0f);
            C[(size_t)m * N + n] = v;
        }
    }
}

// ---- Grouped (MoE-style) GEMM over the per-sample indexed weights ----------
// Rows bucketed into 16-row tiles sharing block index g. Padded lanes read
// row 0 (their output row is simply not stored -- rows are independent).
// Fixed MAXTILES grid; early-exit on device-side tile count (static grid for
// graph capture). K = EDIM = 512 always.
template<int NT>
__global__ __launch_bounds__(32)
void grouped_gemm_kernel(const float* __restrict__ Act, const float* __restrict__ W_all,
                         const float* __restrict__ b_all, float* __restrict__ Out,
                         const int* __restrict__ tile_group, const int* __restrict__ tile_rows,
                         const int* __restrict__ n_tiles, int N, int relu) {
    int t = blockIdx.x;
    if (t >= *n_tiles) return;
    int lane = threadIdx.x;
    int g = tile_group[t];
    int n0 = blockIdx.y * 16 * NT;
    int nl = lane & 15;
    const float* W = W_all + (size_t)g * EDIM * N;
    const float* bias = b_all + (size_t)g * N;
    int my_row = tile_rows[t * 16 + nl];
    int row_eff = my_row < 0 ? 0 : my_row;
    const float* Arow = Act + (size_t)row_eff * EDIM;

    v8f acc[NT];
#pragma unroll
    for (int tt = 0; tt < NT; ++tt) acc[tt] = {};

    for (int k0 = 0; k0 < EDIM; k0 += 32) {
        if (k0 + 32 < EDIM)
            __builtin_prefetch(W + (size_t)(k0 + 32) * N + n0, 0, 1);
        v16bf a = load_a_frag(Arow, k0, lane);
        v16bf b[NT];
#pragma unroll
        for (int tt = 0; tt < NT; ++tt)
            b[tt] = load_b_frag(W, N, k0, n0 + tt * 16 + nl, lane);
#pragma unroll
        for (int tt = 0; tt < NT; ++tt)
            acc[tt] = wmma_bf16(a, b[tt], acc[tt]);
    }
#pragma unroll
    for (int tt = 0; tt < NT; ++tt) {
        int n = n0 + tt * 16 + nl;
        float bv = bias[n];
#pragma unroll
        for (int i = 0; i < 8; ++i) {
            int mrow = tile_rows[t * 16 + (lane >> 4) * 8 + i];
            float v = acc[tt][i] + bv;
            if (relu) v = fmaxf(v, 0.0f);
            if (mrow >= 0) Out[(size_t)mrow * N + n] = v;
        }
    }
}

// ---- Zero-pad x [512,784] -> x_pad [512,800] -------------------------------
__global__ void pad_x_kernel(const float* __restrict__ x, float* __restrict__ x_pad) {
    int i = blockIdx.x * blockDim.x + threadIdx.x;
    if (i >= BATCH * IN_PAD) return;
    int r = i / IN_PAD, c = i - r * IN_PAD;
    x_pad[i] = (c < IN_DIM) ? x[(size_t)r * IN_DIM + c] : 0.0f;
}

// ---- Zero-pad emb_W [784,512] -> embW_pad [800,512] (row-major, same N) ----
__global__ void pad_w_kernel(const float* __restrict__ w, float* __restrict__ w_pad) {
    int i = blockIdx.x * blockDim.x + threadIdx.x;
    if (i >= IN_PAD * EDIM) return;
    w_pad[i] = (i < IN_DIM * EDIM) ? w[i] : 0.0f;
}

// ---- Bucket samples by block index into padded 16-row tiles ----------------
__global__ __launch_bounds__(BATCH)
void bucket_kernel(const int* __restrict__ flatidx, int* __restrict__ tile_group,
                   int* __restrict__ tile_rows, int* __restrict__ n_tiles) {
    __shared__ int cnt[NBLK];
    __shared__ int tile_base[NBLK];
    __shared__ int mypos[BATCH];
    int tid = threadIdx.x;
    if (tid < NBLK) cnt[tid] = 0;
    __syncthreads();
    int g = flatidx[tid];
    mypos[tid] = atomicAdd(&cnt[g], 1);
    __syncthreads();
    if (tid == 0) {
        int tb = 0;
        for (int i = 0; i < NBLK; ++i) {
            tile_base[i] = tb;
            int tg = (cnt[i] + 15) >> 4;
            for (int t = 0; t < tg; ++t) tile_group[tb + t] = i;
            tb += tg;
        }
        *n_tiles = tb;
    }
    __syncthreads();
    for (int i = tid; i < MAXTILES * 16; i += BATCH) tile_rows[i] = -1;
    __syncthreads();
    int slot = mypos[tid];
    int tile = tile_base[g] + (slot >> 4);
    tile_rows[tile * 16 + (slot & 15)] = tid;
}

// ---- Per-sample L2 norm (one wave32 per sample) ----------------------------
__global__ __launch_bounds__(32)
void norm_kernel(const float* __restrict__ state, float* __restrict__ norms) {
    int b = blockIdx.x;
    int lane = threadIdx.x;
    float s = 0.0f;
    for (int e = lane; e < EDIM; e += 32) {
        float v = state[(size_t)b * EDIM + e];
        s += v * v;
    }
#pragma unroll
    for (int off = 16; off > 0; off >>= 1) s += __shfl_xor(s, off, 32);
    if (lane == 0) norms[b] = sqrtf(s);
}

__global__ void normalize_kernel(const float* __restrict__ traw, const float* __restrict__ norms,
                                 float* __restrict__ t, int n) {
    int i = blockIdx.x * blockDim.x + threadIdx.x;
    if (i < n) {
        int b = i / EDIM;
        t[i] = traw[i] / (norms[b] + 1e-6f);
    }
}

__global__ void residual_kernel(const float* __restrict__ t, const float* __restrict__ initial,
                                float* __restrict__ state, float ramp, int n) {
    int i = blockIdx.x * blockDim.x + threadIdx.x;
    if (i < n) state[i] = t[i] + ramp * initial[i];
}

// argmax over logits.reshape(B, 2, 8) axis=2 -> flat = a0*8 + a1 (first-max wins)
__global__ void argmax_kernel(const float* __restrict__ logits, int* __restrict__ flatidx) {
    int b = blockIdx.x * blockDim.x + threadIdx.x;
    if (b >= BATCH) return;
    const float* L = logits + (size_t)b * ABSZ;
    int i0 = 0, i1 = 0;
    float m0 = L[0], m1 = L[8];
#pragma unroll
    for (int i = 1; i < 8; ++i) {
        if (L[i] > m0)     { m0 = L[i];     i0 = i; }
        if (L[8 + i] > m1) { m1 = L[8 + i]; i1 = i; }
    }
    flatidx[b] = i0 * 8 + i1;
}

__global__ void zero_idx_kernel(int* __restrict__ flatidx) {
    int i = blockIdx.x * blockDim.x + threadIdx.x;
    if (i < BATCH) flatidx[i] = 0;
}

// ---------------------------------------------------------------------------

extern "C" void kernel_launch(void* const* d_in, const int* in_sizes, int n_in,
                              void* d_out, int out_size, void* d_ws, size_t ws_size,
                              hipStream_t stream) {
    (void)in_sizes; (void)n_in; (void)out_size; (void)ws_size;

    const float* x      = (const float*)d_in[0];
    const float* emb_W  = (const float*)d_in[1];
    const float* emb_b  = (const float*)d_in[2];
    const float* st_W1  = (const float*)d_in[3];
    const float* st_b1  = (const float*)d_in[4];
    const float* st_W2  = (const float*)d_in[5];
    const float* st_b2  = (const float*)d_in[6];
    const float* ad_W1  = (const float*)d_in[7];
    const float* ad_b1  = (const float*)d_in[8];
    const float* ad_W2  = (const float*)d_in[9];
    const float* ad_b2  = (const float*)d_in[10];
    const float* out_W1 = (const float*)d_in[11];
    const float* out_b1 = (const float*)d_in[12];
    const float* out_W2 = (const float*)d_in[13];
    const float* out_b2 = (const float*)d_in[14];
    float* out = (float*)d_out;

    // workspace layout (~8.5 MB)
    float* initial  = (float*)d_ws;
    float* state    = initial  + (size_t)BATCH * EDIM;
    float* hbuf     = state    + (size_t)BATCH * EDIM;
    float* traw     = hbuf     + (size_t)BATCH * EDIM;
    float* tbuf     = traw     + (size_t)BATCH * EDIM;
    float* x_pad    = tbuf     + (size_t)BATCH * EDIM;      // [512, 800]
    float* embW_pad = x_pad    + (size_t)BATCH * IN_PAD;    // [800, 512]
    float* logits   = embW_pad + (size_t)IN_PAD * EDIM;
    float* norms    = logits   + (size_t)BATCH * ABSZ;
    int*   flatidx    = (int*)(norms + BATCH);
    int*   tile_group = flatidx + BATCH;
    int*   tile_rows  = tile_group + MAXTILES;
    int*   n_tiles    = tile_rows + MAXTILES * 16;

    const dim3 w(32);
    const int NELE = BATCH * EDIM;
    const int EB = 256;
    const int EG = (NELE + EB - 1) / EB;

    // zero-pad embed inputs so all GEMMs have K % 32 == 0 (no tail paths)
    pad_x_kernel<<<(BATCH * IN_PAD + EB - 1) / EB, EB, 0, stream>>>(x, x_pad);
    pad_w_kernel<<<(IN_PAD * EDIM + EB - 1) / EB, EB, 0, stream>>>(emb_W, embW_pad);

    // state = initial = x @ emb_W + emb_b
    dense_gemm_kernel<4><<<dim3(BATCH / 16, EDIM / 64), w, 0, stream>>>(
        x_pad, embW_pad, emb_b, initial, EDIM, IN_PAD, 0);
    hipMemcpyAsync(state, initial, (size_t)NELE * sizeof(float),
                   hipMemcpyDeviceToDevice, stream);

    // initial address: next_address(initial_state, idx=0)
    zero_idx_kernel<<<2, 256, 0, stream>>>(flatidx);
    bucket_kernel<<<1, BATCH, 0, stream>>>(flatidx, tile_group, tile_rows, n_tiles);
    grouped_gemm_kernel<4><<<dim3(MAXTILES, EDIM / 64), w, 0, stream>>>(
        state, ad_W1, ad_b1, hbuf, tile_group, tile_rows, n_tiles, EDIM, 1);
    grouped_gemm_kernel<1><<<dim3(MAXTILES, 1), w, 0, stream>>>(
        hbuf, ad_W2, ad_b2, logits, tile_group, tile_rows, n_tiles, ABSZ, 0);
    argmax_kernel<<<2, 256, 0, stream>>>(logits, flatidx);

    for (int i = 0; i < JUMPS; ++i) {
        float ramp = (float)i / (float)(JUMPS - 1);
        bucket_kernel<<<1, BATCH, 0, stream>>>(flatidx, tile_group, tile_rows, n_tiles);
        // through_block
        grouped_gemm_kernel<4><<<dim3(MAXTILES, EDIM / 64), w, 0, stream>>>(
            state, st_W1, st_b1, hbuf, tile_group, tile_rows, n_tiles, EDIM, 1);
        grouped_gemm_kernel<4><<<dim3(MAXTILES, EDIM / 64), w, 0, stream>>>(
            hbuf, st_W2, st_b2, traw, tile_group, tile_rows, n_tiles, EDIM, 1);
        norm_kernel<<<BATCH, 32, 0, stream>>>(state, norms);
        normalize_kernel<<<EG, EB, 0, stream>>>(traw, norms, tbuf, NELE);
        // next_address(t, idx)
        grouped_gemm_kernel<4><<<dim3(MAXTILES, EDIM / 64), w, 0, stream>>>(
            tbuf, ad_W1, ad_b1, hbuf, tile_group, tile_rows, n_tiles, EDIM, 1);
        grouped_gemm_kernel<1><<<dim3(MAXTILES, 1), w, 0, stream>>>(
            hbuf, ad_W2, ad_b2, logits, tile_group, tile_rows, n_tiles, ABSZ, 0);
        argmax_kernel<<<2, 256, 0, stream>>>(logits, flatidx);
        // state = t + ramp * initial
        residual_kernel<<<EG, EB, 0, stream>>>(tbuf, initial, state, ramp, NELE);
    }

    // final = through_block(state, idx) + initial
    bucket_kernel<<<1, BATCH, 0, stream>>>(flatidx, tile_group, tile_rows, n_tiles);
    grouped_gemm_kernel<4><<<dim3(MAXTILES, EDIM / 64), w, 0, stream>>>(
        state, st_W1, st_b1, hbuf, tile_group, tile_rows, n_tiles, EDIM, 1);
    grouped_gemm_kernel<4><<<dim3(MAXTILES, EDIM / 64), w, 0, stream>>>(
        hbuf, st_W2, st_b2, traw, tile_group, tile_rows, n_tiles, EDIM, 1);
    norm_kernel<<<BATCH, 32, 0, stream>>>(state, norms);
    normalize_kernel<<<EG, EB, 0, stream>>>(traw, norms, tbuf, NELE);
    residual_kernel<<<EG, EB, 0, stream>>>(tbuf, initial, state, 1.0f, NELE);

    // output head
    dense_gemm_kernel<4><<<dim3(BATCH / 16, EDIM / 64), w, 0, stream>>>(
        state, out_W1, out_b1, hbuf, EDIM, EDIM, 1);
    dense_gemm_kernel<4><<<dim3(BATCH / 16, NC / 64), w, 0, stream>>>(
        hbuf, out_W2, out_b2, out, NC, EDIM, 0);
}
